// GATv2EncoderLayer_17291538334012
// MI455X (gfx1250) — compile-verified, matrix operand order checked by reading, not run on hardware
//
#include <hip/hip_runtime.h>
#include <math.h>

#define NSLOPE 0.2f

typedef __attribute__((ext_vector_type(16))) _Float16 v16h;
typedef __attribute__((ext_vector_type(8)))  _Float16 v8h;
typedef __attribute__((ext_vector_type(8)))  float    v8f;

// ---- WMMA fragment layout (CDNA5 16x16x32 f16, wave32) ----
// A (16x32): lane m=lane&15; lanes 0-15 hold K={0..7,16..23}, lanes 16-31 K={8..15,24..31}
//   element e -> K = kbase + (e&7) + ((e>>3)<<4),  kbase = (lane&16)?8:0
// B (32x16): mirror with N = lane&15.
// C (f32 16x16): VGPR j -> row j (lanes 0-15) / j+8 (lanes 16-31), col = lane&15.
// Packed operand buffers store fragments contiguously: frag f, lane l -> 16 halves
// at [(f*32 + l)*16].

// load A fragment from row-major fp32 (used only where transpose is unavoidable)
static __device__ __forceinline__ v16h load_A_rm(const float* __restrict__ src,
                                                 int row0, int ld, int k0, int lane) {
  int m  = lane & 15;
  int kb = k0 + ((lane & 16) ? 8 : 0);
  const float* p = src + (size_t)(row0 + m) * ld + kb;
  v16h a;
#pragma unroll
  for (int e = 0; e < 16; ++e) {
    int k = (e & 7) + ((e >> 3) << 4);
    a[e] = (_Float16)p[k];
  }
  return a;
}

static __device__ __forceinline__ v16h load_frag(const _Float16* __restrict__ base,
                                                 size_t frag, int lane) {
  return *(const v16h*)(base + (frag * 32 + lane) * 16);
}

static __device__ __forceinline__ void store_C_rm(float* __restrict__ dst, v8f c,
                                                  int row0, int ld, int n0, int lane,
                                                  const float* __restrict__ bias) {
  int n  = n0 + (lane & 15);
  int rb = row0 + ((lane & 16) ? 8 : 0);
  float badd = bias ? bias[n] : 0.0f;
#pragma unroll
  for (int j = 0; j < 8; ++j)
    dst[(size_t)(rb + j) * ld + n] = c[j] + badd;
}

static __device__ __forceinline__ float gelu_exact(float x) {
  return 0.5f * x * (1.0f + erff(x * 0.70710678118654752f));
}

static __device__ __forceinline__ void atomicMaxF(float* addr, float val) {
  int cur = __float_as_int(*addr);
  while (__int_as_float(cur) < val) {
    int prev = atomicCAS((int*)addr, cur, __float_as_int(val));
    if (prev == cur) break;
    cur = prev;
  }
}

// ---------------- packing kernels ----------------
// Pack row-major W [K, Ncols] fp32 -> f16 B-fragments, frag index = kt*ntiles + nt.
__global__ void k_packW(const float* __restrict__ W, _Float16* __restrict__ out,
                        int K, int Ncols) {
  int t = blockIdx.x * blockDim.x + threadIdx.x;
  int ntiles = Ncols >> 4, ktiles = K >> 5;
  if (t >= ktiles * ntiles * 32) return;
  int lane = t & 31;
  int nt = (t >> 5) % ntiles;
  int kt = (t >> 5) / ntiles;
  int n  = nt * 16 + (lane & 15);
  int kb = kt * 32 + ((lane & 16) ? 8 : 0);
  _Float16* o = out + (size_t)t * 16;
#pragma unroll
  for (int e = 0; e < 16; ++e) {
    int k = kb + (e & 7) + ((e >> 3) << 4);
    o[e] = (_Float16)W[(size_t)k * Ncols + n];
  }
}

// Pack row-major src [rows, ld] fp32 -> f16 A-fragments, frag index = rt*ktiles + kt.
__global__ void k_packA(const float* __restrict__ src, _Float16* __restrict__ out,
                        int ld, int rowtiles) {
  int t = blockIdx.x * blockDim.x + threadIdx.x;
  int ktiles = ld >> 5;
  if (t >= rowtiles * ktiles * 32) return;
  int lane = t & 31;
  int kt = (t >> 5) % ktiles;
  int rt = (t >> 5) / ktiles;
  int m  = lane & 15;
  int kb = kt * 32 + ((lane & 16) ? 8 : 0);
  const float* p = src + (size_t)(rt * 16 + m) * ld + kb;
  _Float16* o = out + (size_t)t * 16;
#pragma unroll
  for (int e = 0; e < 16; ++e) {
    int k = (e & 7) + ((e >> 3) << 4);
    o[e] = (_Float16)p[k];
  }
}

// ---------------- init / mean ----------------
__global__ void k_init(float* nodemax, float* denom, float* meanacc, int n4) {
  int t = blockIdx.x * blockDim.x + threadIdx.x;
  if (t < n4) { nodemax[t] = -3.0e38f; denom[t] = 0.0f; }
  if (blockIdx.x == 0 && t < 32) meanacc[t] = 0.0f;
}

__global__ void k_mean_acc(const float* __restrict__ ea, float* meanacc,
                           int E, int nthreads) {
  int t  = blockIdx.x * blockDim.x + threadIdx.x;
  int c  = t & 31;
  int r0 = t >> 5;
  int rs = nthreads >> 5;
  float s = 0.f;
  for (int r = r0; r < E; r += rs) s += ea[(size_t)r * 32 + c];
  atomicAdd(&meanacc[c], s);
}

__global__ void k_mean_fin(float* meanacc, float invE) {
  if (threadIdx.x < 32) meanacc[threadIdx.x] *= invE;
}

// ---------------- node projections: xl, xr, agg=x@Wres+bias_gat ----------------
__global__ void __launch_bounds__(96) k_proj(const _Float16* __restrict__ xp,
    const _Float16* __restrict__ Wlp, const _Float16* __restrict__ Wrp,
    const _Float16* __restrict__ Wresp,
    const float* __restrict__ bl, const float* __restrict__ br,
    const float* __restrict__ bgat,
    float* __restrict__ xl, float* __restrict__ xr, float* __restrict__ agg) {
  int lane = threadIdx.x & 31;
  int wid  = threadIdx.x >> 5;
  int rt   = blockIdx.x;
  int row0 = rt * 16;
  const _Float16* Wp; const float* bias; float* out;
  if (wid == 0)      { Wp = Wlp;   bias = bl;   out = xl;  }
  else if (wid == 1) { Wp = Wrp;   bias = br;   out = xr;  }
  else               { Wp = Wresp; bias = bgat; out = agg; }
  v16h A[4];
#pragma unroll
  for (int kt = 0; kt < 4; ++kt) A[kt] = load_frag(xp, (size_t)rt * 4 + kt, lane);
#pragma unroll
  for (int nt = 0; nt < 16; ++nt) {
    v8f acc = {};
#pragma unroll
    for (int kt = 0; kt < 4; ++kt)
      acc = __builtin_amdgcn_wmma_f32_16x16x32_f16(
          false, A[kt], false, load_frag(Wp, (size_t)kt * 16 + nt, lane),
          (short)0, acc, false, false);
    store_C_rm(out, acc, row0, 256, nt * 16, lane, bias);
  }
}

// ---------------- edge pass A: logits + segment max ----------------
__global__ void k_edgeA(const long long* __restrict__ ei,
                        const float* __restrict__ ea, const float* __restrict__ meanea,
                        const float* __restrict__ We, const float* __restrict__ att,
                        const float* __restrict__ xl, const float* __restrict__ xr,
                        float* __restrict__ logits, float* nodemax,
                        int E, int Etot) {
  int lane = threadIdx.x & 31;
  long long e = (long long)blockIdx.x * (blockDim.x >> 5) + (threadIdx.x >> 5);
  if (e >= Etot) return;
  long long src, dst; float av;
  if (e < E) { src = ei[e]; dst = ei[E + e]; av = ea[(size_t)e * 32 + lane]; }
  else       { src = dst = e - E;            av = meanea[lane]; }
  const float* xls = xl + (size_t)src * 256;
  const float* xrd = xr + (size_t)dst * 256;
  int c0 = lane * 8;               // head h = lane>>3 (64 ch/head = 8 lanes)
  float s[8];
#pragma unroll
  for (int i = 0; i < 8; ++i) s[i] = xls[c0 + i] + xrd[c0 + i];
  // eproj = ea_row @ We, recomputed (saves 870MB materialization)
#pragma unroll 4
  for (int k = 0; k < 32; ++k) {
    float ak = __shfl(av, k, 32);
    const float* wrow = We + (size_t)k * 256 + c0;
#pragma unroll
    for (int i = 0; i < 8; ++i) s[i] = fmaf(ak, wrow[i], s[i]);
  }
  float l = 0.f;
#pragma unroll
  for (int i = 0; i < 8; ++i) {
    float v = s[i];
    v = v > 0.f ? v : NSLOPE * v;
    l = fmaf(v, att[c0 + i], l);
  }
  l += __shfl_xor(l, 1, 32);
  l += __shfl_xor(l, 2, 32);
  l += __shfl_xor(l, 4, 32);
  if ((lane & 7) == 0) {
    int h = lane >> 3;
    logits[e * 4 + h] = l;
    atomicMaxF(&nodemax[dst * 4 + h], l);
  }
}

// ---------------- edge pass B: alpha + segment sum of denominators ----------------
__global__ void k_edgeB(const long long* __restrict__ ei, float* __restrict__ alpha,
                        const float* __restrict__ nodemax, float* denom,
                        int E, int Etot) {
  long long t = (long long)blockIdx.x * blockDim.x + threadIdx.x;
  if (t >= (long long)Etot * 4) return;
  long long e = t >> 2; int h = (int)(t & 3);
  long long dst = (e < E) ? ei[E + e] : (e - E);
  float a = __expf(alpha[t] - nodemax[dst * 4 + h]);
  alpha[t] = a;
  atomicAdd(&denom[dst * 4 + h], a);
}

// ---------------- edge pass C: message scatter ----------------
__global__ void k_edgeC(const long long* __restrict__ ei,
                        const float* __restrict__ alpha, const float* __restrict__ denom,
                        const float* __restrict__ xl, float* agg, int E, int Etot) {
  int lane = threadIdx.x & 31;
  long long e = (long long)blockIdx.x * (blockDim.x >> 5) + (threadIdx.x >> 5);
  if (e >= Etot) return;
  long long src, dst;
  if (e < E) { src = ei[e]; dst = ei[E + e]; } else src = dst = e - E;
  int h = lane >> 3;
  float w = alpha[e * 4 + h] / denom[dst * 4 + h];
  int c0 = lane * 8;
  const float* xls = xl + (size_t)src * 256 + c0;
  float* ag = agg + (size_t)dst * 256 + c0;
#pragma unroll
  for (int i = 0; i < 8; ++i) atomicAdd(&ag[i], xls[i] * w);
}

// ---------------- LN1 + GELU per node, output directly in packed A-frag order ----
__global__ void k_ln1(const float* __restrict__ agg, const float* __restrict__ g1,
                      const float* __restrict__ be1, _Float16* __restrict__ act1p,
                      int N) {
  int lane = threadIdx.x & 31;
  long long n = (long long)blockIdx.x * (blockDim.x >> 5) + (threadIdx.x >> 5);
  if (n >= N) return;
  int c0 = lane * 8;
  const float* row = agg + (size_t)n * 256;
  float v[8]; float s = 0.f, q = 0.f;
#pragma unroll
  for (int i = 0; i < 8; ++i) { v[i] = row[c0 + i]; s += v[i]; q += v[i] * v[i]; }
#pragma unroll
  for (int m = 1; m < 32; m <<= 1) { s += __shfl_xor(s, m, 32); q += __shfl_xor(q, m, 32); }
  float mu  = s * (1.f / 256.f);
  float var = q * (1.f / 256.f) - mu * mu;
  float inv = rsqrtf(var + 1e-5f);
  v8h o;
#pragma unroll
  for (int i = 0; i < 8; ++i) {
    float t = (v[i] - mu) * inv * g1[c0 + i] + be1[c0 + i];
    o[i] = (_Float16)gelu_exact(t);
  }
  // channel c = c0+i -> frag kt=c>>5 (= lane>>2), kin0=(lane&3)*8
  long long rt = n >> 4;  int mr = (int)(n & 15);
  int kt    = lane >> 2;
  int kin0  = (lane & 3) * 8;
  int flane = mr + ((kin0 & 8) ? 16 : 0);
  int ebase = (kin0 & 16) ? 8 : 0;
  *(v8h*)(act1p + ((rt * 8 + kt) * 32 + flane) * 16 + ebase) = o;
}

// ---------------- heads transform GEMM [N,256]x[256,64] ----------------
// stores h2 fp32 (residual) and h2p packed f16 A-fragments (for MLP GEMM1)
__global__ void __launch_bounds__(128) k_heads(const _Float16* __restrict__ act1p,
    const _Float16* __restrict__ Whp, const float* __restrict__ bh,
    float* __restrict__ h2, _Float16* __restrict__ h2p) {
  int lane = threadIdx.x & 31;
  int nt   = threadIdx.x >> 5;
  int rt   = blockIdx.x;
  int row0 = rt * 16;
  v8f acc = {};
#pragma unroll
  for (int kt = 0; kt < 8; ++kt)
    acc = __builtin_amdgcn_wmma_f32_16x16x32_f16(
        false, load_frag(act1p, (size_t)rt * 8 + kt, lane),
        false, load_frag(Whp, (size_t)kt * 4 + nt, lane),
        (short)0, acc, false, false);
  int n   = nt * 16 + (lane & 15);
  int rb  = (lane & 16) ? 8 : 0;
  float b = bh[n];
  int kt  = n >> 5;             // packed A-frag coords for channel n
  int kin = n & 31;
  int e   = (kin & 7) + ((kin & 16) ? 8 : 0);
  int flx = (kin & 8) ? 16 : 0;
  _Float16* hp = h2p + ((size_t)(rt * 2 + kt) * 32) * 16;
#pragma unroll
  for (int j = 0; j < 8; ++j) {
    float v = acc[j] + b;
    int r = rb + j;
    h2[(size_t)(row0 + r) * 64 + n] = v;
    hp[(size_t)(r + flx) * 16 + e] = (_Float16)v;
  }
}

// ---------------- fused MLP (WMMA) + residual + LN2 ----------------
__global__ void __launch_bounds__(128) k_mlp(const float* __restrict__ h2,
    const _Float16* __restrict__ h2p,
    const _Float16* __restrict__ W1p, const float* __restrict__ b1,
    const _Float16* __restrict__ W2p, const float* __restrict__ b2,
    const float* __restrict__ g2, const float* __restrict__ be2,
    float* __restrict__ out, int rowtiles) {
  __shared__ float lds[4][16 * 64];
  int lane = threadIdx.x & 31;
  int wid  = threadIdx.x >> 5;
  int rt   = blockIdx.x * 4 + wid;
  if (rt >= rowtiles) rt = rowtiles - 1;   // clamp: keep EXEC full for WMMA
  int row0 = rt * 16;
  float* L = lds[wid];

  // t = gelu(h2 @ W1 + b1) -> LDS (C-layout to row-major transpose via LDS)
  v16h A[2];
#pragma unroll
  for (int kt = 0; kt < 2; ++kt) A[kt] = load_frag(h2p, (size_t)rt * 2 + kt, lane);
#pragma unroll
  for (int nt = 0; nt < 4; ++nt) {
    v8f acc = {};
#pragma unroll
    for (int kt = 0; kt < 2; ++kt)
      acc = __builtin_amdgcn_wmma_f32_16x16x32_f16(
          false, A[kt], false, load_frag(W1p, (size_t)kt * 4 + nt, lane),
          (short)0, acc, false, false);
    int n  = nt * 16 + (lane & 15);
    int rb = (lane & 16) ? 8 : 0;
    float bb = b1[n];
#pragma unroll
    for (int j = 0; j < 8; ++j)
      L[(rb + j) * 64 + n] = gelu_exact(acc[j] + bb);
  }

  // out_pre = t @ W2 + b2 + h2 (residual), written back to LDS
#pragma unroll
  for (int kt = 0; kt < 2; ++kt) A[kt] = load_A_rm(L, 0, 64, kt * 32, lane);
#pragma unroll
  for (int nt = 0; nt < 4; ++nt) {
    v8f acc = {};
#pragma unroll
    for (int kt = 0; kt < 2; ++kt)
      acc = __builtin_amdgcn_wmma_f32_16x16x32_f16(
          false, A[kt], false, load_frag(W2p, (size_t)kt * 4 + nt, lane),
          (short)0, acc, false, false);
    int n  = nt * 16 + (lane & 15);
    int rb = (lane & 16) ? 8 : 0;
    float bb = b2[n];
#pragma unroll
    for (int j = 0; j < 8; ++j)
      L[(rb + j) * 64 + n] = acc[j] + bb + h2[(size_t)(row0 + rb + j) * 64 + n];
  }

  // LN2 over 64 per row: 2 lanes per row, combine with shfl_xor(16)
  int r    = lane & 15;
  int half = lane >> 4;
  float s = 0.f, q = 0.f;
#pragma unroll
  for (int i = 0; i < 32; ++i) { float v = L[r * 64 + half * 32 + i]; s += v; q += v * v; }
  s += __shfl_xor(s, 16, 32);
  q += __shfl_xor(q, 16, 32);
  float mu  = s * (1.f / 64.f);
  float var = q * (1.f / 64.f) - mu * mu;
  float inv = rsqrtf(var + 1e-5f);
#pragma unroll
  for (int i = 0; i < 32; ++i) {
    int cc = half * 32 + i;
    float v = L[r * 64 + cc];
    out[(size_t)(row0 + r) * 64 + cc] = (v - mu) * inv * g2[cc] + be2[cc];
  }
}

// ---------------- host ----------------
extern "C" void kernel_launch(void* const* d_in, const int* in_sizes, int n_in,
                              void* d_out, int out_size, void* d_ws, size_t ws_size,
                              hipStream_t stream) {
  const float*     x    = (const float*)d_in[0];
  const long long* ei   = (const long long*)d_in[1];
  const float*     ea   = (const float*)d_in[2];
  const float*     Wl   = (const float*)d_in[3];
  const float*     bl   = (const float*)d_in[4];
  const float*     Wr   = (const float*)d_in[5];
  const float*     br   = (const float*)d_in[6];
  const float*     We   = (const float*)d_in[7];
  const float*     att  = (const float*)d_in[8];
  const float*     Wres = (const float*)d_in[9];
  const float*     bgat = (const float*)d_in[10];
  const float*     Wh   = (const float*)d_in[11];
  const float*     bh   = (const float*)d_in[12];
  const float*     W1   = (const float*)d_in[13];
  const float*     b1   = (const float*)d_in[14];
  const float*     W2   = (const float*)d_in[15];
  const float*     b2   = (const float*)d_in[16];
  const float*     g1   = (const float*)d_in[17];
  const float*     be1  = (const float*)d_in[18];
  const float*     g2   = (const float*)d_in[19];
  const float*     be2  = (const float*)d_in[20];
  (void)n_in; (void)out_size; (void)ws_size;

  const int N        = in_sizes[0] / 128;
  const int E        = in_sizes[1] / 2;
  const int Etot     = E + N;
  const int rowtiles = (N + 15) / 16;

  char* ws = (char*)d_ws;
  size_t off = 0;
  auto alloc = [&](size_t bytes) {
    void* p = ws + off;
    off += (bytes + 255) & ~(size_t)255;
    return p;
  };
  float* xl      = (float*)alloc((size_t)N * 256 * 4);
  float* xr      = (float*)alloc((size_t)N * 256 * 4);
  float* agg     = (float*)alloc((size_t)N * 256 * 4);
  float* h2      = (float*)alloc((size_t)N * 64 * 4);
  float* logits  = (float*)alloc((size_t)Etot * 4 * 4);
  float* nodemax = (float*)alloc((size_t)N * 4 * 4);
  float* denom   = (float*)alloc((size_t)N * 4 * 4);
  float* meanacc = (float*)alloc(32 * 4);
  _Float16* xp    = (_Float16*)alloc((size_t)N * 128 * 2);
  _Float16* act1p = (_Float16*)alloc((size_t)N * 256 * 2);
  _Float16* h2p   = (_Float16*)alloc((size_t)N * 64 * 2);
  _Float16* Wlp   = (_Float16*)alloc((size_t)128 * 256 * 2);
  _Float16* Wrp   = (_Float16*)alloc((size_t)128 * 256 * 2);
  _Float16* Wresp = (_Float16*)alloc((size_t)128 * 256 * 2);
  _Float16* Whp   = (_Float16*)alloc((size_t)256 * 64 * 2);
  _Float16* W1p   = (_Float16*)alloc((size_t)64 * 64 * 2);
  _Float16* W2p   = (_Float16*)alloc((size_t)64 * 64 * 2);

  int n4 = N * 4;
  k_init<<<(n4 + 255) / 256, 256, 0, stream>>>(nodemax, denom, meanacc, n4);
  k_mean_acc<<<128, 256, 0, stream>>>(ea, meanacc, E, 128 * 256);
  k_mean_fin<<<1, 32, 0, stream>>>(meanacc, 1.0f / (float)E);

  // fragment packing (one-shot, tiny vs. edge-phase traffic)
  int tA = rowtiles * 4 * 32;
  k_packA<<<(tA + 255) / 256, 256, 0, stream>>>(x, xp, 128, rowtiles);
  k_packW<<<8, 256, 0, stream>>>(Wl, Wlp, 128, 256);
  k_packW<<<8, 256, 0, stream>>>(Wr, Wrp, 128, 256);
  k_packW<<<8, 256, 0, stream>>>(Wres, Wresp, 128, 256);
  k_packW<<<4, 256, 0, stream>>>(Wh, Whp, 256, 64);
  k_packW<<<1, 256, 0, stream>>>(W1, W1p, 64, 64);
  k_packW<<<1, 256, 0, stream>>>(W2, W2p, 64, 64);

  k_proj<<<rowtiles, 96, 0, stream>>>(xp, Wlp, Wrp, Wresp, bl, br, bgat, xl, xr, agg);

  int eblocks = (Etot + 7) / 8;   // 8 waves per 256-thread block, 1 wave/edge
  k_edgeA<<<eblocks, 256, 0, stream>>>(ei, ea, meanacc, We, att, xl, xr,
                                       logits, nodemax, E, Etot);
  long long tB = (long long)Etot * 4;
  k_edgeB<<<(int)((tB + 255) / 256), 256, 0, stream>>>(ei, logits, nodemax, denom, E, Etot);
  k_edgeC<<<eblocks, 256, 0, stream>>>(ei, logits, denom, xl, agg, E, Etot);

  k_ln1<<<(N + 7) / 8, 256, 0, stream>>>(agg, g1, be1, act1p, N);
  k_heads<<<rowtiles, 128, 0, stream>>>(act1p, Whp, bh, h2, h2p);
  k_mlp<<<(rowtiles + 3) / 4, 128, 0, stream>>>(h2, h2p, W1p, b1, W2p, b2, g2, be2,
                                                (float*)d_out, rowtiles);
}
